// CharacteristicFeatureComputer_54855322304543
// MI455X (gfx1250) — matrix operand order repaired and checked by reading, not compiled, not run on hardware
//
#include <hip/hip_runtime.h>

typedef __attribute__((ext_vector_type(16))) _Float16 v16h;
typedef __attribute__((ext_vector_type(8)))  float    v8f;

#define Q_TOT   4096      // nt*nx
#define K_P     32
#define H_DIM   32
#define D_IN    85
#define WAVES   4         // waves per block
#define NTILES  32768     // (B*Q*K)/16 row-tiles
#define PI_F    3.14159265358979323846f

// ds_swizzle_b32 group-of-32 xor patterns: offset = xor_mask<<10 | and_mask(0x1f)
#define SWZ_XOR1 0x041F
#define SWZ_XOR2 0x081F
#define SWZ_XOR4 0x101F
#define SWZ_XOR8 0x201F
#define SWZ_ADD(v, imm) \
    ((v) += __int_as_float(__builtin_amdgcn_ds_swizzle(__float_as_int(v), (imm))))

// swizzle helpers for the 16-bit WMMA A-matrix layout (wave32):
//   lane = M + 16*half ; element j of v16h holds K = (j&7) + 8*half + 16*(j>=8)
// inverse: given K (0..31): half = (K>>3)&1 ; j = (K&7) | ((K>>4)<<3)
__device__ __host__ constexpr int a_half(int k) { return (k >> 3) & 1; }
__device__ __host__ constexpr int a_elem(int k) { return (k & 7) | ((k >> 4) << 3); }
// element offset (within one wave's a_lds block) of column `col`, excluding row*16
__device__ __host__ constexpr int a_addr(int col) {
    return (((col >> 5) << 5) + (a_half(col & 31) << 4)) * 16 + a_elem(col & 31);
}

// Branch-free erf (Abramowitz-Stegun 7.1.26, |err| <= 1.5e-7).
__device__ __forceinline__ float erf_bf(float x) {
    float ax = __builtin_fabsf(x);
    float t  = __builtin_amdgcn_rcpf(__builtin_fmaf(0.3275911f, ax, 1.0f));
    float p  = __builtin_fmaf(t, 1.061405429f, -1.453152027f);
    p = __builtin_fmaf(t, p, 1.421413741f);
    p = __builtin_fmaf(t, p, -0.284496736f);
    p = __builtin_fmaf(t, p, 0.254829592f);
    p = p * t;
    float e  = __expf(-ax * ax);
    float r  = __builtin_fmaf(-p, e, 1.0f);
    return __builtin_copysignf(r, x);
}

__device__ __forceinline__ float gelu_exact(float x) {
    return 0.5f * x * (1.0f + erf_bf(x * 0.70710678118654752f));
}

__global__ __launch_bounds__(128) void cfc_fused_kernel(
    const float* __restrict__ t_coords, const float* __restrict__ x_coords,
    const float* __restrict__ xs_p,     const float* __restrict__ ks_p,
    const float* __restrict__ mask_p,
    const float* __restrict__ W1p, const float* __restrict__ b1p,
    const float* __restrict__ lgp, const float* __restrict__ lbp,
    const float* __restrict__ W2p, const float* __restrict__ b2p,
    float* __restrict__ out)
{
    __shared__ __align__(16) _Float16 a_lds[WAVES][3 * 32 * 16];   // 12 KB
    __shared__ __align__(16) _Float16 w1_lds[2 * 3 * 512];          // 6 KB
    __shared__ __align__(16) _Float16 w2_lds[2 * 512];              // 2 KB
    __shared__ __align__(16) _Float16 h_lds[WAVES][512];            // 4 KB
    __shared__ float base_lds[WAVES][80];                           // 1.25 KB

    const int tid  = threadIdx.x;
    const int lane = tid & 31;
    const int wave = tid >> 5;
    const int hf   = lane >> 4;      // wave half (0/1)
    const int lnid = lane & 15;

    // ---------------- stage W1 / W2 into B-fragment layout (whole block) ----
    // B layout (matches A): elem j -> K = (j&7) + 8*half + 16*(j>=8)
    for (int i = tid; i < 2 * 3 * 512; i += 128) {
        int j  = i & 15;
        int l  = (i >> 4) & 31;
        int kb = (i >> 9) % 3;
        int nt = i / 1536;
        int k  = (j & 7) + 8 * (l >> 4) + 16 * (j >> 3) + kb * 32;
        int n  = nt * 16 + (l & 15);
        w1_lds[i] = (k < D_IN) ? (_Float16)W1p[k * H_DIM + n] : (_Float16)0.0f;
    }
    for (int i = tid; i < 2 * 512; i += 128) {
        int j  = i & 15;
        int l  = (i >> 4) & 31;
        int nt = i >> 9;
        int k  = (j & 7) + 8 * (l >> 4) + 16 * (j >> 3);
        int n  = nt * 16 + (l & 15);
        w2_lds[i] = (_Float16)W2p[k * H_DIM + n];
    }

    // ---------------- per-wave tile identity --------------------------------
    const int tile = blockIdx.x * WAVES + wave;   // 0 .. NTILES-1
    const int bq   = tile >> 1;                   // flat (b*Q + q)
    const int k0   = (tile & 1) << 4;             // piece offset (0 or 16)
    const int bb   = bq >> 12;                    // bq / Q_TOT
    const float t  = t_coords[bq];
    const float x  = x_coords[bq];

    // ---------------- base features (16 rows x 5 feats = 80 tasks) ----------
    {
        int nidx = 2 + (lane < 16 ? 1 : 0);       // 80 = 2*32 + 16
        for (int i = 0; i < nidx; ++i) {
            int idx = lane + 32 * i;
            int row = idx / 5, f = idx - row * 5;
            int kk  = k0 + row;
            float xl  = xs_p[bb * (K_P + 1) + kk];
            float xr  = xs_p[bb * (K_P + 1) + kk + 1];
            float lam = ks_p[bb * K_P + kk];
            float xc  = 0.5f * (xl + xr);
            float ts  = fmaxf(t, 0.001f);
            float v;
            if      (f == 0) v = (x - xc) / ts;
            else if (f == 1) v = x - xc - lam * t;
            else if (f == 2) v = x - (xl + lam * t);
            else if (f == 3) v = x - (xr + lam * t);
            else             v = t;               // t_feat: not clamped
            if (f < 4) v = fminf(fmaxf(v, -10.0f), 10.0f);
            base_lds[wave][idx] = v;
        }
    }
    // zero the padded K-block (cols 85..95 live in kb=2) with 128-bit stores
    {
        uint4 z = make_uint4(0u, 0u, 0u, 0u);
        uint4* zp = (uint4*)&a_lds[wave][2 * 512];  // 1024 B
        zp[lane]      = z;
        zp[lane + 32] = z;
    }
    __syncthreads();

    // ---------------- fourier encode -> swizzled A layout --------------------
    // lane owns (row = lnid). half 0 emits sin terms (sub 1..8), half 1 the
    // cos terms (sub 9..16) via cos(w*v) = sin(w*v + pi/2): one v_sin path,
    // no divergence. All columns are compile-time constants after unrolling,
    // so each store is [row*16 + const-select] -> near-zero address VALU.
    {
        const int   rb    = lnid << 4;
        const float phase = hf ? 1.57079632679489662f : 0.0f;
        #pragma unroll
        for (int f = 0; f < 5; ++f) {
            float v = base_lds[wave][lnid * 5 + f];
            // base feature (sub=0): both halves write the same value/address
            a_lds[wave][a_addr(f * 17) + rb] = (_Float16)v;
            #pragma unroll
            for (int e = 0; e < 8; ++e) {
                float val = __sinf(__builtin_fmaf(v, PI_F * (float)(1 << e), phase));
                const int c0 = f * 17 + 1 + e;          // half0 col; half1 = c0+8
                int ad = hf ? a_addr(c0 + 8) : a_addr(c0);
                a_lds[wave][ad + rb] = (_Float16)val;
            }
        }
    }
    __syncthreads();

    // ---------------- GEMM1: [16 x 96] @ [96 x 32] via 6 WMMAs --------------
    v8f acc0 = {}, acc1 = {};
    #pragma unroll
    for (int kb = 0; kb < 3; ++kb) {
        v16h a  = *(const v16h*)&a_lds[wave][((kb << 5) + lane) << 4];
        v16h w0 = *(const v16h*)&w1_lds[(0 * 3 + kb) * 512 + (lane << 4)];
        v16h w1 = *(const v16h*)&w1_lds[(1 * 3 + kb) * 512 + (lane << 4)];
        acc0 = __builtin_amdgcn_wmma_f32_16x16x32_f16(false, a, false, w0,
                                                      (short)0, acc0, false, false);
        acc1 = __builtin_amdgcn_wmma_f32_16x16x32_f16(false, a, false, w1,
                                                      (short)0, acc1, false, false);
    }

    // ---------------- bias + exact GELU (branch-free erf) -------------------
    const int n0 = lnid, n1 = 16 + lnid;
    const float bia0 = b1p[n0], bia1 = b1p[n1];
    float g0[8], g1[8];
    #pragma unroll
    for (int r = 0; r < 8; ++r) {
        g0[r] = gelu_exact(acc0[r] + bia0);
        g1[r] = gelu_exact(acc1[r] + bia1);
    }

    // ---------------- LayerNorm over H=32 (row = r + 8*hf) -------------------
    // ds_swizzle xor-reductions stay within each 16-lane group on wave32.
    const float lg0 = lgp[n0], lg1 = lgp[n1], lb0 = lbp[n0], lb1 = lbp[n1];
    const int hj0 = (a_half(n0) << 8) + a_elem(n0);   // n0 part of h_lds addr
    const int hj1 = (a_half(n1) << 8) + a_elem(n1);
    #pragma unroll
    for (int r = 0; r < 8; ++r) {
        float s = g0[r] + g1[r];
        float q = g0[r] * g0[r] + g1[r] * g1[r];
        SWZ_ADD(s, SWZ_XOR1); SWZ_ADD(s, SWZ_XOR2);
        SWZ_ADD(s, SWZ_XOR4); SWZ_ADD(s, SWZ_XOR8);
        SWZ_ADD(q, SWZ_XOR1); SWZ_ADD(q, SWZ_XOR2);
        SWZ_ADD(q, SWZ_XOR4); SWZ_ADD(q, SWZ_XOR8);
        float mu  = s * (1.0f / 32.0f);
        float var = __builtin_fmaf(-mu, mu, q * (1.0f / 32.0f));
        float inv = __builtin_amdgcn_rsqf(var + 1e-5f);   // raw v_rsq_f32
        float y0  = (g0[r] - mu) * inv * lg0 + lb0;
        float y1  = (g1[r] - mu) * inv * lg1 + lb1;
        int Mt = (r + (hf << 3)) << 4;                    // M*16
        h_lds[wave][hj0 + Mt] = (_Float16)y0;
        h_lds[wave][hj1 + Mt] = (_Float16)y1;
    }
    __syncthreads();

    // ---------------- GEMM2: [16 x 32] @ [32 x 32] via 2 WMMAs ---------------
    v16h ah  = *(const v16h*)&h_lds[wave][lane << 4];
    v16h wb0 = *(const v16h*)&w2_lds[0 * 512 + (lane << 4)];
    v16h wb1 = *(const v16h*)&w2_lds[1 * 512 + (lane << 4)];
    v8f o0 = {}, o1 = {};
    o0 = __builtin_amdgcn_wmma_f32_16x16x32_f16(false, ah, false, wb0,
                                                (short)0, o0, false, false);
    o1 = __builtin_amdgcn_wmma_f32_16x16x32_f16(false, ah, false, wb1,
                                                (short)0, o1, false, false);

    // ---------------- bias + mask + store ------------------------------------
    const float bo0 = b2p[n0], bo1 = b2p[n1];
    const int R0 = tile << 4;
    #pragma unroll
    for (int r = 0; r < 8; ++r) {
        int M = r + (hf << 3);
        float m = mask_p[bb * K_P + k0 + M];
        int R = R0 + M;
        out[R * H_DIM + n0] = (o0[r] + bo0) * m;
        out[R * H_DIM + n1] = (o1[r] + bo1) * m;
    }
}

extern "C" void kernel_launch(void* const* d_in, const int* in_sizes, int n_in,
                              void* d_out, int out_size, void* d_ws, size_t ws_size,
                              hipStream_t stream) {
    const float* t_coords = (const float*)d_in[0];
    const float* x_coords = (const float*)d_in[1];
    const float* xs       = (const float*)d_in[2];
    const float* ks       = (const float*)d_in[3];
    const float* mask     = (const float*)d_in[4];
    const float* W1       = (const float*)d_in[5];
    const float* b1       = (const float*)d_in[6];
    const float* ln_g     = (const float*)d_in[7];
    const float* ln_b     = (const float*)d_in[8];
    const float* W2       = (const float*)d_in[9];
    const float* b2       = (const float*)d_in[10];
    float* out = (float*)d_out;

    dim3 grid(NTILES / WAVES);   // 8192 blocks
    dim3 block(WAVES * 32);      // 128 threads = 4 waves
    cfc_fused_kernel<<<grid, block, 0, stream>>>(
        t_coords, x_coords, xs, ks, mask, W1, b1, ln_g, ln_b, W2, b2, out);
}
